// GPTJAttention_56109452755192
// MI455X (gfx1250) — matrix operand order repaired and checked by reading, not compiled, not run on hardware
//
#include <hip/hip_runtime.h>
#include <hip/hip_bf16.h>

// ---------------------------------------------------------------------------
// GPT-J decode attention for MI455X (gfx1250, wave32, WMMA).
// B=16,S=1,P=2048,D=4096,H=16,HD=256,ROT=64
// Bandwidth-bound: ~2.4 GB traffic -> ~103 us floor at 23.3 TB/s.
// ---------------------------------------------------------------------------

#define BB   16
#define SS   1
#define PP   2048
#define DD   4096
#define HH   16
#define HD   256
#define ROTD 64
#define KVLEN (PP + SS)   // 2049

typedef __attribute__((ext_vector_type(16))) __bf16 v16bf;
typedef __attribute__((ext_vector_type(8)))  float  v8f;

// ---------------------------------------------------------------------------
// One K=32 step of the 16x16 WMMA tile: C += A[16x32] * B[32x16] (bf16 in,
// f32 acc).  A and B both use contiguous k = 16*(lane/16) + s.
// ---------------------------------------------------------------------------
__device__ inline v8f wmma_k32_step(const float* __restrict__ arow,
                                    const float* __restrict__ wrow,
                                    int d0, int kbase, int ksel, v8f c)
{
    const float4* pa0 = (const float4*)(arow + d0 + ksel);
    const float4* pa1 = (const float4*)(arow + d0 + 8 + ksel);
    float4 a0 = pa0[0], a1 = pa0[1];
    float4 a2 = pa1[0], a3 = pa1[1];
    const float4* pb  = (const float4*)(wrow + d0 + ksel);
    float4 b0 = pb[0], b1 = pb[1], b2 = pb[2], b3 = pb[3];

    float af[16] = {a0.x,a0.y,a0.z,a0.w, a1.x,a1.y,a1.z,a1.w,
                    a2.x,a2.y,a2.z,a2.w, a3.x,a3.y,a3.z,a3.w};
    float bf[16] = {b0.x,b0.y,b0.z,b0.w, b1.x,b1.y,b1.z,b1.w,
                    b2.x,b2.y,b2.z,b2.w, b3.x,b3.y,b3.z,b3.w};
    v16bf av, bv;
    #pragma unroll
    for (int j = 0; j < 16; ++j) { av[j] = (__bf16)af[j]; bv[j] = (__bf16)bf[j]; }

    return __builtin_amdgcn_wmma_f32_16x16x32_bf16(
               false, av, false, bv, (short)0, c, false, false);
}

// ---------------------------------------------------------------------------
// Wave-level 16x16 tile of C(16 x 4096) = A(16 x 4096) * W^T,
// out[m,n] = sum_d A[m,d] * W[n,d].  Dual accumulators break the
// WMMA->WMMA RAW chain (5-cycle hazard class for bf16) and let two
// iterations' loads overlap.
// ---------------------------------------------------------------------------
__device__ inline v8f wmma_tile_rowblock(const float* __restrict__ A,
                                         const float* __restrict__ W,
                                         int n0, int lane)
{
    const int m     = lane & 15;
    const int hi    = lane >> 4;
    const int kbase = hi ? 8 : 0;    // unused: A frag uses ksel too (lanes 16-31: K=16..31)
    const int ksel  = hi ? 16 : 0;   // B frag: lanes 16-31 hold K=16..31

    const float* __restrict__ arow = A + (size_t)m * DD;
    const float* __restrict__ wrow = W + (size_t)(n0 + m) * DD;  // (lane&15)==m

    v8f c0 = {}, c1 = {};
    for (int d0 = 0; d0 < DD; d0 += 64) {
        c0 = wmma_k32_step(arow, wrow, d0,      kbase, ksel, c0);
        c1 = wmma_k32_step(arow, wrow, d0 + 32, kbase, ksel, c1);
    }
    return c0 + c1;
}

// C-fragment writeback: VGPR j -> row (j + hi*8), col n0 + (lane&15)
__device__ inline void wmma_tile_store(v8f c, float* __restrict__ out,
                                       int n0, int lane)
{
    const int hi = lane >> 4;
    const int n  = n0 + (lane & 15);
    #pragma unroll
    for (int j = 0; j < 8; ++j) {
        int mj = j + (hi ? 8 : 0);
        out[(size_t)mj * DD + n] = c[j];
    }
}

// ---------------------------------------------------------------------------
// Kernel 1: q/k/v projections.  3 matrices x 256 N-tiles = 768 wave-tiles.
// grid 192 blocks x 128 threads (4 waves/block).
// ---------------------------------------------------------------------------
__global__ __launch_bounds__(128) void k_qkv_proj(
    const float* __restrict__ hidden,
    const float* __restrict__ wq,
    const float* __restrict__ wk,
    const float* __restrict__ wv,
    float* __restrict__ qraw,   // (B, D)
    float* __restrict__ kraw,
    float* __restrict__ vraw)
{
    const int lane = threadIdx.x & 31;
    const int wid  = blockIdx.x * 4 + (threadIdx.x >> 5);   // 0..767
    const int mat  = wid >> 8;                              // 0=q 1=k 2=v
    const int n0   = (wid & 255) * 16;

    const float* W  = (mat == 0) ? wq : (mat == 1) ? wk : wv;
    float* out      = (mat == 0) ? qraw : (mat == 1) ? kraw : vraw;

    v8f c = wmma_tile_rowblock(hidden, W, n0, lane);
    wmma_tile_store(c, out, n0, lane);
}

// ---------------------------------------------------------------------------
// Kernel 2: GPT-J interleaved RoPE on q,k; append new k/v to cache tails.
// ---------------------------------------------------------------------------
__global__ __launch_bounds__(256) void k_rope_append(
    const float* __restrict__ qraw,
    const float* __restrict__ kraw,
    const float* __restrict__ vraw,
    const int*   __restrict__ pos_ids,    // (B, S)
    float* __restrict__ qrope,            // (B, H, HD) == (B, D)
    float* __restrict__ out_k,            // (B, H, KVLEN, HD)
    float* __restrict__ out_v)
{
    const int idx = blockIdx.x * 256 + threadIdx.x;   // 0..65535
    const int b   = idx >> 12;
    const int e   = idx & (DD - 1);
    const int h   = e >> 8;
    const int hd  = e & (HD - 1);

    const float pos = (float)pos_ids[b];
    float qv = qraw[idx];
    float kv = kraw[idx];
    float qo = qv, ko = kv;
    if (hd < ROTD) {
        const int i2    = hd & ~1;
        const float fr  = __powf(10000.0f, -(float)i2 / (float)ROTD);
        const float ang = pos * fr;
        const float s   = __sinf(ang);
        const float c   = __cosf(ang);
        const float qrot = (hd & 1) ? qraw[idx - 1] : -qraw[idx + 1];
        const float krot = (hd & 1) ? kraw[idx - 1] : -kraw[idx + 1];
        qo = qv * c + qrot * s;
        ko = kv * c + krot * s;
    }
    qrope[idx] = qo;
    const size_t tail = ((size_t)b * HH + h) * (size_t)KVLEN * HD + (size_t)PP * HD + hd;
    out_k[tail] = ko;
    out_v[tail] = vraw[idx];
}

// ---------------------------------------------------------------------------
// Kernel 3: fused attention + KV-cache copy.  One block (256 thr) per (b,h).
// Phase A: wave-cooperative scores (each wave reads a full 1KB key row with
//          2x b128 per lane) while streaming the row out to k_full.
// Phase B: key-partitioned PV: thread owns a float4 of channels, the block's
//          four 64-thread groups interleave keys, all b128 traffic; partial
//          accumulators combined through LDS.
// ---------------------------------------------------------------------------
__global__ __launch_bounds__(256) void k_attention(
    const float* __restrict__ past_k,   // (B, H, P, HD)
    const float* __restrict__ past_v,
    const float* __restrict__ qrope,    // (B, H, HD)
    const float* __restrict__ mask,     // (B, 1, 1, KVLEN)
    float* __restrict__ out_k,          // (B, H, KVLEN, HD)  tail pre-filled
    float* __restrict__ out_v,
    float* __restrict__ ctx)            // (B, H, HD)
{
    __shared__ float q_s[HD];
    __shared__ float sc[2064];          // KVLEN=2049, padded
    __shared__ float red[16];
    __shared__ float vred[4][HD / 4][4];  // 4 key-partitions x 64 ch-groups x f4

    const int bh   = blockIdx.x;        // b*H + h
    const int b    = bh >> 4;
    const int tid  = threadIdx.x;
    const int lane = tid & 31;
    const int wid  = tid >> 5;

    if (tid < HD) q_s[tid] = qrope[(size_t)bh * HD + tid];
    __syncthreads();

    const float* pk = past_k + (size_t)bh * PP * HD;
    float*       ok = out_k  + (size_t)bh * KVLEN * HD;

    const float4 q0 = *(const float4*)&q_s[lane * 8];
    const float4 q1 = *(const float4*)&q_s[lane * 8 + 4];

    // ---- Phase A: scores + K copy ----
    for (int kk = wid; kk < KVLEN; kk += 8) {
        const float* krow = (kk < PP) ? (pk + (size_t)kk * HD)
                                      : (ok + (size_t)PP * HD);
        __builtin_prefetch(pk + (size_t)(kk + 8) * HD + lane * 8, 0, 0);
        const float4 x0 = ((const float4*)(krow + lane * 8))[0];
        const float4 x1 = ((const float4*)(krow + lane * 8))[1];
        if (kk < PP) {
            float4* dk = (float4*)(ok + (size_t)kk * HD + lane * 8);
            dk[0] = x0; dk[1] = x1;
        }
        float s = x0.x * q0.x + x0.y * q0.y + x0.z * q0.z + x0.w * q0.w
                + x1.x * q1.x + x1.y * q1.y + x1.z * q1.z + x1.w * q1.w;
        #pragma unroll
        for (int off = 16; off > 0; off >>= 1) s += __shfl_xor(s, off, 32);
        if (lane == 0)
            sc[kk] = s * (1.0f / 16.0f) + mask[(size_t)b * KVLEN + kk];
    }
    __syncthreads();

    // ---- softmax over KVLEN (all keys visible: q_pos = P) ----
    float lm = -3.402823e38f;
    for (int i = tid; i < KVLEN; i += 256) lm = fmaxf(lm, sc[i]);
    #pragma unroll
    for (int off = 16; off > 0; off >>= 1) lm = fmaxf(lm, __shfl_xor(lm, off, 32));
    if (lane == 0) red[wid] = lm;
    __syncthreads();
    float gm = red[0];
    #pragma unroll
    for (int w = 1; w < 8; ++w) gm = fmaxf(gm, red[w]);

    float ls = 0.0f;
    for (int i = tid; i < KVLEN; i += 256) {
        float e = __expf(sc[i] - gm);
        sc[i] = e;
        ls += e;
    }
    #pragma unroll
    for (int off = 16; off > 0; off >>= 1) ls += __shfl_xor(ls, off, 32);
    if (lane == 0) red[8 + wid] = ls;
    __syncthreads();
    float gs = 0.0f;
    #pragma unroll
    for (int w = 0; w < 8; ++w) gs += red[8 + w];
    const float inv = 1.0f / gs;

    // ---- Phase B: ctx = P @ V fused with V copy, all-b128 ----
    const float* pv = past_v + (size_t)bh * PP * HD;
    float*       ov = out_v  + (size_t)bh * KVLEN * HD;

    const int part = tid >> 6;          // key partition 0..3
    const int cg   = tid & 63;          // channel group: floats 4*cg..4*cg+3
    float4 acc = make_float4(0.f, 0.f, 0.f, 0.f);

    #pragma unroll 4
    for (int kk = part; kk < PP; kk += 4) {
        const float4 vv = *(const float4*)(pv + (size_t)kk * HD + cg * 4);
        *(float4*)(ov + (size_t)kk * HD + cg * 4) = vv;
        const float p = sc[kk];
        acc.x = fmaf(p, vv.x, acc.x);
        acc.y = fmaf(p, vv.y, acc.y);
        acc.z = fmaf(p, vv.z, acc.z);
        acc.w = fmaf(p, vv.w, acc.w);
    }
    if (part == 0) {                    // new token's V row (tail, pre-filled)
        const float4 vv = *(const float4*)(ov + (size_t)PP * HD + cg * 4);
        const float p = sc[PP];
        acc.x = fmaf(p, vv.x, acc.x);
        acc.y = fmaf(p, vv.y, acc.y);
        acc.z = fmaf(p, vv.z, acc.z);
        acc.w = fmaf(p, vv.w, acc.w);
    }
    vred[part][cg][0] = acc.x;
    vred[part][cg][1] = acc.y;
    vred[part][cg][2] = acc.z;
    vred[part][cg][3] = acc.w;
    __syncthreads();

    if (tid < 64) {
        float4 r;
        r.x = (vred[0][tid][0] + vred[1][tid][0]) + (vred[2][tid][0] + vred[3][tid][0]);
        r.y = (vred[0][tid][1] + vred[1][tid][1]) + (vred[2][tid][1] + vred[3][tid][1]);
        r.z = (vred[0][tid][2] + vred[1][tid][2]) + (vred[2][tid][2] + vred[3][tid][2]);
        r.w = (vred[0][tid][3] + vred[1][tid][3]) + (vred[2][tid][3] + vred[3][tid][3]);
        r.x *= inv; r.y *= inv; r.z *= inv; r.w *= inv;
        *(float4*)(ctx + (size_t)bh * HD + tid * 4) = r;
    }
}

// ---------------------------------------------------------------------------
// Kernel 4: output projection attn_out(16x4096) = ctx(16x4096) * wo^T
// ---------------------------------------------------------------------------
__global__ __launch_bounds__(128) void k_out_proj(
    const float* __restrict__ ctx,
    const float* __restrict__ wo,
    float* __restrict__ out_attn)
{
    const int lane = threadIdx.x & 31;
    const int wid  = blockIdx.x * 4 + (threadIdx.x >> 5);  // 0..255
    const int n0   = wid * 16;
    v8f c = wmma_tile_rowblock(ctx, wo, n0, lane);
    wmma_tile_store(c, out_attn, n0, lane);
}

// ---------------------------------------------------------------------------
extern "C" void kernel_launch(void* const* d_in, const int* in_sizes, int n_in,
                              void* d_out, int out_size, void* d_ws, size_t ws_size,
                              hipStream_t stream)
{
    const float* hidden  = (const float*)d_in[0];
    const float* past_k  = (const float*)d_in[1];
    const float* past_v  = (const float*)d_in[2];
    const float* wq      = (const float*)d_in[3];
    const float* wk      = (const float*)d_in[4];
    const float* wv      = (const float*)d_in[5];
    const float* wo      = (const float*)d_in[6];
    const float* mask    = (const float*)d_in[7];
    const int*   pos_ids = (const int*)d_in[8];

    const size_t ATTN_ELEMS = (size_t)BB * SS * DD;               // 65536
    const size_t KV_ELEMS   = (size_t)BB * HH * KVLEN * HD;       // 134283264
    float* out_attn = (float*)d_out;
    float* out_k    = out_attn + ATTN_ELEMS;
    float* out_v    = out_k + KV_ELEMS;

    float* ws    = (float*)d_ws;
    float* qraw  = ws;
    float* kraw  = ws + 1 * ATTN_ELEMS;
    float* vraw  = ws + 2 * ATTN_ELEMS;
    float* qrope = ws + 3 * ATTN_ELEMS;
    float* ctx   = ws + 4 * ATTN_ELEMS;

    k_qkv_proj<<<dim3(192), dim3(128), 0, stream>>>(
        hidden, wq, wk, wv, qraw, kraw, vraw);

    k_rope_append<<<dim3(256), dim3(256), 0, stream>>>(
        qraw, kraw, vraw, pos_ids, qrope, out_k, out_v);

    k_attention<<<dim3(BB * HH), dim3(256), 0, stream>>>(
        past_k, past_v, qrope, mask, out_k, out_v, ctx);

    k_out_proj<<<dim3(64), dim3(128), 0, stream>>>(
        ctx, wo, out_attn);
}
